// AttentionModule_39591008534921
// MI455X (gfx1250) — matrix operand order repaired and checked by reading, not compiled, not run on hardware
//
#include <hip/hip_runtime.h>
#include <hip/hip_bf16.h>
#include <math.h>

// ---------------------------------------------------------------------------
// AttentionModule for MI455X (gfx1250, wave32, WMMA)
//
// reference:  scores = tanh(cat(hid,enc) @ Wa.T + ba) @ Wa2.T  (+ba2, dropped:
//             softmax shift-invariant);  w = softmax_S(scores);
//             applied = sum_s w * enc;  out = tanh(cat(dec,applied) @ Wc.T + bc)
//
// Pipeline:
//   k_convert_We : W_attn[:,H:2H] f32 -> bf16 workspace (4 MiB, L2-resident)
//   k_hidpart    : hid_part[b,n] = b_attn[n] + hidden[b,:] . W_attn[n,:H]
//   k_attn_scores: bf16 WMMA GEMM enc@We.T fused with +hid_part, tanh, .Wa2
//   k_softmax_ap : softmax over S, applied = einsum("bs,sbh->bh")
//   k_combine    : out = tanh(cat(dec,applied) @ Wc.T + bc)
// ---------------------------------------------------------------------------

typedef __attribute__((ext_vector_type(16))) __bf16 v16bf;
typedef __attribute__((ext_vector_type(8)))  float  v8f;

#define S_DIM 512
#define B_DIM 64
#define H_DIM 1024
#define N_DIM 2048                  // 2H
#define M_DIM (S_DIM * B_DIM)       // 32768 rows, m = s*64 + b

// main-kernel tiling
#define M_TILE   128                // rows per block (8 waves x 16)
#define A_STRIDE 1032               // 1024 + 8 bf16 pad (16B-aligned rows, bank-skewed)
#define KC       512                // K chunk staged for B per step
#define B_STRIDE 520                // 512 + 8 pad
#define NT_W     32                 // N columns per outer step (2 wmma tiles / wave)

union FragBF { v16bf v; uint4 q[2]; };
union Pack4  { __bf16 h[4]; uint2 u; };

__device__ __forceinline__ float fast_tanh(float x) {
  // tanh(x) = 1 - 2/(exp(2x)+1); exp->inf gives 1, exp->0 gives -1 (no NaN)
  const float e = __expf(2.0f * x);
  return 1.0f - 2.0f / (e + 1.0f);
}

__device__ __forceinline__ float dot1024(const float4* __restrict__ a,
                                         const float4* __restrict__ w) {
  float acc = 0.0f;
#pragma unroll 8
  for (int i = 0; i < 256; ++i) {
    const float4 x = a[i], y = w[i];
    acc += x.x * y.x + x.y * y.y + x.z * y.z + x.w * y.w;
  }
  return acc;
}

// --------------------------- prep kernels ----------------------------------

__global__ __launch_bounds__(256) void k_convert_We(const float* __restrict__ Wa,
                                                    __bf16* __restrict__ We) {
  const int idx = blockIdx.x * 256 + threadIdx.x;      // 0 .. 2048*1024-1
  const int n = idx >> 10, k = idx & 1023;
  We[idx] = (__bf16)Wa[(size_t)n * N_DIM + H_DIM + k]; // encoder half of W_attn
}

__global__ __launch_bounds__(256) void k_hidpart(const float* __restrict__ hidden,
                                                 const float* __restrict__ Wa,
                                                 const float* __restrict__ ba,
                                                 float* __restrict__ hidp) {
  const int idx = blockIdx.x * 256 + threadIdx.x;      // 0 .. 64*2048-1
  const int b = idx >> 11, n = idx & 2047;
  const float acc = dot1024((const float4*)(hidden + (size_t)b * H_DIM),
                            (const float4*)(Wa + (size_t)n * N_DIM));
  hidp[idx] = ba[n] + acc;                              // hid_part[b][n]
}

// --------------------- fused WMMA score kernel -----------------------------
// scores[s*64+b] = sum_n tanh( (enc_row . We[n]) + hid_part[b][n] ) * Wa2[n]

__global__ __launch_bounds__(256, 1)
void k_attn_scores(const float* __restrict__ enc,    // [M_DIM][H_DIM] f32
                   const __bf16* __restrict__ We,    // [N_DIM][H_DIM] bf16
                   const float* __restrict__ hidp,   // [B_DIM][N_DIM] f32
                   const float* __restrict__ Wa2,    // [N_DIM] f32
                   float* __restrict__ scores)       // [M_DIM] f32
{
  extern __shared__ char smem[];
  __bf16* Alds = reinterpret_cast<__bf16*>(smem);                                 // [128][A_STRIDE]
  __bf16* Blds = reinterpret_cast<__bf16*>(smem + (size_t)M_TILE * A_STRIDE * 2); // [32][B_STRIDE]

  const int tid  = threadIdx.x;
  const int wave = tid >> 5;
  const int lane = tid & 31;
  const int half = lane >> 4;           // K-half selector (ISA fragment layout)
  const int nl   = lane & 15;           // row-in-tile (A) / column-in-tile (B,C)
  const int mBase   = blockIdx.x * M_TILE;
  const int rowBase = wave * 16;        // wave's 16 M-rows within the block tile

  // ---- stage A tile once: 128 x 1024 f32 -> bf16 LDS (coalesced float4) ----
  // H_DIM/4 == 256 -> use shifts, no div/mod temporaries.
  for (int i = tid; i < M_TILE * 256; i += 256) {
    const int r = i >> 8;
    const int c = (i & 255) << 2;
    const float4 f = *reinterpret_cast<const float4*>(enc + (size_t)(mBase + r) * H_DIM + c);
    Pack4 p;
    p.h[0] = (__bf16)f.x; p.h[1] = (__bf16)f.y;
    p.h[2] = (__bf16)f.z; p.h[3] = (__bf16)f.w;
    *reinterpret_cast<uint2*>(Alds + (size_t)r * A_STRIDE + c) = p.u;
  }

  float scoreAcc[8];
#pragma unroll
  for (int r = 0; r < 8; ++r) scoreAcc[r] = 0.0f;

  // Precompute per-lane LDS fragment bases (constant across the nt loop).
  const __bf16* Arow  = Alds + (size_t)(rowBase + nl) * A_STRIDE + half * 8;
  const __bf16* Brow0 = Blds + (size_t)nl * B_STRIDE + half * 8;
  const __bf16* Brow1 = Blds + (size_t)(16 + nl) * B_STRIDE + half * 8;

  for (int nt = 0; nt < N_DIM / NT_W; ++nt) {
    const int n0 = nt * NT_W;
    v8f acc0 = {}, acc1 = {};

    // hint-prefetch next N-tile's B rows into L2 (global_prefetch_b8)
    {
      const size_t nb = ((size_t)(n0 + NT_W) * H_DIM) & ((size_t)N_DIM * H_DIM - 1);
      __builtin_prefetch(We + nb + (size_t)tid * 64, 0, 1);
    }

    for (int kc = 0; kc < H_DIM; kc += KC) {
      __syncthreads();  // A staged (first pass) / previous B consumed
      // ---- stage B: 32 rows x KC bf16, coalesced uint4 (KC/8 == 64) ----
      for (int i = tid; i < NT_W * 64; i += 256) {
        const int r = i >> 6;
        const int c = (i & 63) << 3;
        *reinterpret_cast<uint4*>(Blds + (size_t)r * B_STRIDE + c) =
            *reinterpret_cast<const uint4*>(We + (size_t)(n0 + r) * H_DIM + kc + c);
      }
      __syncthreads();

      // ISA fragment layout (16-bit A 16x32): lanes 0-15 K={0..7,16..23},
      // lanes 16-31 K={8..15,24..31}; B 32x16: lane%16 = column, same K split as A.
      // Partial unroll (4) keeps fragment live-ranges short -> no scratch spills.
#pragma unroll 4
      for (int j = 0; j < KC / 32; ++j) {
        const int kb = kc + j * 32;   // global K for A
        const int lb = j * 32;        // local K for B
        FragBF a, b0, b1;
        a.q[0]  = *reinterpret_cast<const uint4*>(Arow + kb);
        a.q[1]  = *reinterpret_cast<const uint4*>(Arow + kb + 16);
        b0.q[0] = *reinterpret_cast<const uint4*>(Brow0 + lb);
        b0.q[1] = *reinterpret_cast<const uint4*>(Brow0 + lb + 16);
        b1.q[0] = *reinterpret_cast<const uint4*>(Brow1 + lb);
        b1.q[1] = *reinterpret_cast<const uint4*>(Brow1 + lb + 16);
        acc0 = __builtin_amdgcn_wmma_f32_16x16x32_bf16(false, a.v, false, b0.v,
                                                       (short)0, acc0, false, false);
        acc1 = __builtin_amdgcn_wmma_f32_16x16x32_bf16(false, a.v, false, b1.v,
                                                       (short)0, acc1, false, false);
      }
    }

    // ---- fused epilogue: +hid_part, tanh, dot with Wa2, per-lane accumulate.
    // C layout: VGPR r -> (M = r + half*8, N = n-tile col nl)
    const int nCol0 = n0 + nl, nCol1 = n0 + 16 + nl;
    const float w20 = Wa2[nCol0], w21 = Wa2[nCol1];
#pragma unroll
    for (int r = 0; r < 8; ++r) {
      const int bIdx = (rowBase + half * 8 + r) & 63;   // m % 64 (mBase % 64 == 0)
      const float* hp = hidp + (size_t)bIdx * N_DIM;
      scoreAcc[r] += fast_tanh(acc0[r] + hp[nCol0]) * w20 +
                     fast_tanh(acc1[r] + hp[nCol1]) * w21;
    }
  }

  // ---- final cross-lane reduce over the 16 columns (stays within lane-half) --
#pragma unroll
  for (int r = 0; r < 8; ++r) {
    float v = scoreAcc[r];
    v += __shfl_xor(v, 1);
    v += __shfl_xor(v, 2);
    v += __shfl_xor(v, 4);
    v += __shfl_xor(v, 8);
    if (nl == 0) scores[mBase + rowBase + half * 8 + r] = v;
  }
}

// ------------------- softmax over S + applied ------------------------------

__global__ __launch_bounds__(256)
void k_softmax_apply(const float* __restrict__ scores,   // [S][B]
                     const float* __restrict__ enc,      // [S][B][H]
                     float* __restrict__ applied)        // [B][H]
{
  __shared__ float wgt[S_DIM];
  __shared__ float red[8];
  const int b = blockIdx.x, tid = threadIdx.x;
  const int wave = tid >> 5, lane = tid & 31;

  const float v0 = scores[(size_t)tid * B_DIM + b];
  const float v1 = scores[(size_t)(tid + 256) * B_DIM + b];

  float m = fmaxf(v0, v1);
#pragma unroll
  for (int off = 16; off; off >>= 1) m = fmaxf(m, __shfl_xor(m, off));
  if (lane == 0) red[wave] = m;
  __syncthreads();
  if (tid == 0) {
    float t = red[0];
    for (int i = 1; i < 8; ++i) t = fmaxf(t, red[i]);
    red[0] = t;
  }
  __syncthreads();
  m = red[0];

  const float e0 = __expf(v0 - m), e1 = __expf(v1 - m);
  float s = e0 + e1;
#pragma unroll
  for (int off = 16; off; off >>= 1) s += __shfl_xor(s, off);
  __syncthreads();
  if (lane == 0) red[wave] = s;
  __syncthreads();
  if (tid == 0) {
    float t = 0.0f;
    for (int i = 0; i < 8; ++i) t += red[i];
    red[0] = t;
  }
  __syncthreads();
  const float inv = 1.0f / red[0];
  wgt[tid]       = e0 * inv;
  wgt[tid + 256] = e1 * inv;
  __syncthreads();

  // applied[b][h] = sum_s wgt[s] * enc[s][b][h]; thread owns 4 h (float4)
  const int h0 = tid * 4;
  float4 acc = make_float4(0.f, 0.f, 0.f, 0.f);
  for (int s2 = 0; s2 < S_DIM; ++s2) {
    const float w = wgt[s2];
    const float4 e = *reinterpret_cast<const float4*>(
        enc + ((size_t)s2 * B_DIM + b) * H_DIM + h0);
    acc.x += w * e.x; acc.y += w * e.y; acc.z += w * e.z; acc.w += w * e.w;
  }
  *reinterpret_cast<float4*>(applied + (size_t)b * H_DIM + h0) = acc;
}

// ------------------------- final combine -----------------------------------

__global__ __launch_bounds__(256)
void k_combine(const float* __restrict__ dec,      // [B][H]
               const float* __restrict__ applied,  // [B][H]
               const float* __restrict__ Wc,       // [H][2H]
               const float* __restrict__ bc,       // [H]
               float* __restrict__ out)            // [B][H]
{
  const int idx = blockIdx.x * 256 + threadIdx.x;  // 0 .. 64*1024-1
  const int b = idx >> 10, h = idx & 1023;
  const float4* wrow = (const float4*)(Wc + (size_t)h * N_DIM);
  float acc = bc[h];
  acc += dot1024((const float4*)(dec + (size_t)b * H_DIM),     wrow);
  acc += dot1024((const float4*)(applied + (size_t)b * H_DIM), wrow + 256);
  out[idx] = fast_tanh(acc);
}

// --------------------------- launcher --------------------------------------

extern "C" void kernel_launch(void* const* d_in, const int* in_sizes, int n_in,
                              void* d_out, int out_size, void* d_ws, size_t ws_size,
                              hipStream_t stream) {
  const float* hidden  = (const float*)d_in[0];
  const float* dec     = (const float*)d_in[1];
  const float* enc     = (const float*)d_in[2];
  const float* W_attn  = (const float*)d_in[3];
  const float* b_attn  = (const float*)d_in[4];
  const float* W_attn2 = (const float*)d_in[5];
  // d_in[6] = b_attn2: constant shift before softmax -> provably a no-op
  const float* W_comb  = (const float*)d_in[7];
  const float* b_comb  = (const float*)d_in[8];

  float* out     = (float*)d_out;                 // [64,1024]  (tuple slot 0)
  float* applied = (float*)d_out + B_DIM * H_DIM; // [64,1024]  (tuple slot 1)

  // workspace: We bf16 (4 MiB) | hid_part f32 (512 KiB) | scores f32 (128 KiB)
  char*   ws     = (char*)d_ws;
  __bf16* ws_We  = (__bf16*)ws;
  float*  ws_hid = (float*)(ws + (size_t)N_DIM * H_DIM * sizeof(__bf16));
  float*  ws_sc  = (float*)(ws + (size_t)N_DIM * H_DIM * sizeof(__bf16)
                               + (size_t)B_DIM * N_DIM * sizeof(float));

  k_convert_We<<<(N_DIM * H_DIM) / 256, 256, 0, stream>>>(W_attn, ws_We);
  k_hidpart<<<(B_DIM * N_DIM) / 256, 256, 0, stream>>>(hidden, W_attn, b_attn, ws_hid);

  const size_t shmem = (size_t)M_TILE * A_STRIDE * sizeof(__bf16)
                     + (size_t)NT_W * B_STRIDE * sizeof(__bf16);   // ~297 KiB < 320 KiB
  k_attn_scores<<<M_DIM / M_TILE, 256, shmem, stream>>>(enc, ws_We, ws_hid, W_attn2, ws_sc);

  k_softmax_apply<<<B_DIM, 256, 0, stream>>>(ws_sc, enc, applied);
  k_combine<<<(B_DIM * H_DIM) / 256, 256, 0, stream>>>(dec, applied, W_comb, b_comb, out);
}
